// VQVAE_16217796510395
// MI455X (gfx1250) — matrix operand order repaired and checked by reading, not compiled, not run on hardware
//
#include <hip/hip_runtime.h>

// ---------------------------------------------------------------------------
// VQ-VAE transformer forward for MI455X (gfx1250), wave32 + WMMA bf16.
//
// Roofline: dominant GEMMs (11200x512x2048 etc.) have arithmetic intensity
// >250 FLOP/byte -> matrix-core bound on MI455X (HBM 23.3 TB/s is not the
// limiter). All GEMM-shaped math goes through v_wmma_f32_16x16x32_bf16.
// GEMM: one wave owns a 32x64 tile (8 WMMAs per 32-K step; A-frags reused 4x
// across N, B-frags reused 2x across M).  Fragment loads are float4
// (global_load_b128); scalar gathers use one base pointer + compile-time
// immediate offsets (folded into the VMEM IOFFSET field, no per-load address
// VALU / xcnt stalls).  f32->bf16 packs two values per v_perm_b32.
// Attention is fused (flash style): 10 unmasked 32-key blocks + 1 masked
// tail; P is re-laid C-layout -> A-layout through 1 KB of LDS per wave.
// ---------------------------------------------------------------------------

#define Bb   32
#define Ls   350
#define DINs 6
#define Hs   512
#define NHs  8
#define DHs  64
#define FFs  2048
#define NLs  4
#define Kcb  1024
#define CDs  64
#define BL   (Bb * Ls)   // 11200

typedef __attribute__((ext_vector_type(16))) __bf16 v16bf;
typedef __attribute__((ext_vector_type(8)))  float  v8f;

#define Z8 ((v8f){0.f,0.f,0.f,0.f,0.f,0.f,0.f,0.f})

union BF16x16 { v16bf v; unsigned short u[16]; unsigned int w[8]; };

// Single f32 -> bf16 (round-half-up), for scalar paths.
__device__ __forceinline__ unsigned short f2bf(float f) {
  union { float f; unsigned int i; } x; x.f = f;
  return (unsigned short)((x.i + 0x8000u) >> 16);
}

// Pack two f32 -> {bf16(b)<<16 | bf16(a)} in one v_perm_b32 (round-half-up).
__device__ __forceinline__ unsigned int pkbf(float a, float b) {
  union { float f; unsigned int i; } xa, xb;
  xa.f = a; xb.f = b;
  unsigned int ra = xa.i + 0x8000u;
  unsigned int rb = xb.i + 0x8000u;
  return __builtin_amdgcn_perm(rb, ra, 0x07060302u);  // {rb[31:16], ra[31:16]}
}

// CDNA5 16-bit A-fragment K index for half j (0..15), lane-group g (= lane/16).
__device__ __forceinline__ int kmap(int j, int g) {
  return j + 8 * g + ((j >= 8) ? 8 : 0);
}

__device__ __forceinline__ float4 ld4(const float* p) { return *(const float4*)p; }

// Pack 8 floats (two float4) into fragment dwords [dw, dw+4).
__device__ __forceinline__ void pack8(BF16x16& f, int dw,
                                      const float4& p, const float4& q) {
  f.w[dw + 0] = pkbf(p.x, p.y);
  f.w[dw + 1] = pkbf(p.z, p.w);
  f.w[dw + 2] = pkbf(q.x, q.y);
  f.w[dw + 3] = pkbf(q.z, q.w);
}
__device__ __forceinline__ void pack8s(BF16x16& f, int dw,
                                       const float4& p, const float4& q, float s) {
  f.w[dw + 0] = pkbf(p.x * s, p.y * s);
  f.w[dw + 1] = pkbf(p.z * s, p.w * s);
  f.w[dw + 2] = pkbf(q.x * s, q.y * s);
  f.w[dw + 3] = pkbf(q.z * s, q.w * s);
}

// ---------------------------------------------------------------------------
// Generic GEMM: C[M,N] = A[M,K] x B[K,N] (+bias, optional ReLU, optional
// B transposed i.e. B[N,K]).
// Requirements (true for every call in this network): Kd % 32 == 0,
// N % 64 == 0.  M is clamped per row.  4 waves/block, 32x64 tile per wave.
// ---------------------------------------------------------------------------
__global__ __launch_bounds__(128) void gemm_wmma_kernel(
    const float* __restrict__ A, const float* __restrict__ Bm,
    const float* __restrict__ bias, float* __restrict__ C,
    int M, int N, int Kd, int transB, int relu)
{
  int wave = threadIdx.x >> 5;
  int lane = threadIdx.x & 31;
  int lr = lane & 15, g = lane >> 4;
  int row0 = blockIdx.y * 32;
  int col0 = blockIdx.x * 256 + wave * 64;
  if (col0 >= N) return;          // uniform per wave

  int ar0 = row0 + lr;      if (ar0 >= M) ar0 = M - 1;
  int ar1 = row0 + 16 + lr; if (ar1 >= M) ar1 = M - 1;
  const float* Arow0 = A + (size_t)ar0 * Kd;
  const float* Arow1 = A + (size_t)ar1 * Kd;

  v8f acc[2][4];
#pragma unroll
  for (int i = 0; i < 2; ++i)
#pragma unroll
    for (int t = 0; t < 4; ++t) acc[i][t] = Z8;

  for (int k0 = 0; k0 < Kd; k0 += 32) {
    if (k0 + 64 < Kd) {
      __builtin_prefetch(Arow0 + k0 + 64, 0, 1);                 // global_prefetch_b8
      __builtin_prefetch(Arow1 + k0 + 64, 0, 1);
      if (!transB) __builtin_prefetch(Bm + (size_t)(k0 + 64) * N + col0, 0, 1);
    }
    // A fragments: two contiguous 8-float runs per lane per row tile.
    BF16x16 a[2];
    {
      const float* p0 = Arow0 + k0 + 8 * g;
      pack8(a[0], 0, ld4(p0),      ld4(p0 + 4));
      pack8(a[0], 4, ld4(p0 + 16), ld4(p0 + 20));
      const float* p1 = Arow1 + k0 + 8 * g;
      pack8(a[1], 0, ld4(p1),      ld4(p1 + 4));
      pack8(a[1], 4, ld4(p1 + 16), ld4(p1 + 20));
    }
    int brow = k0 + lane;
    if (!transB) {
      const float* Bp = Bm + (size_t)k0 * N + col0 + lr;
#pragma unroll
      for (int t = 0; t < 4; ++t) {
        BF16x16 bf;
#pragma unroll
        for (int j = 0; j < 16; ++j)
          bf.u[j] = f2bf(Bp[(size_t)kmap(j, g) * N + t * 16]);
        acc[0][t] = __builtin_amdgcn_wmma_f32_16x16x32_bf16(false, a[0].v, false, bf.v,
                                                            (short)0, acc[0][t], false, false);
        acc[1][t] = __builtin_amdgcn_wmma_f32_16x16x32_bf16(false, a[1].v, false, bf.v,
                                                            (short)0, acc[1][t], false, false);
      }
    } else {
      // One base address; t*16*Kd + kmap(j,g) folds into the VMEM immediate offset.
      const float* Bp = Bm + (size_t)(col0 + lr) * Kd + k0;
#pragma unroll
      for (int t = 0; t < 4; ++t) {
        BF16x16 bf;
#pragma unroll
        for (int j = 0; j < 16; ++j)
          bf.u[j] = f2bf(Bp[(size_t)t * 16 * Kd + kmap(j, g)]);
        acc[0][t] = __builtin_amdgcn_wmma_f32_16x16x32_bf16(false, a[0].v, false, bf.v,
                                                            (short)0, acc[0][t], false, false);
        acc[1][t] = __builtin_amdgcn_wmma_f32_16x16x32_bf16(false, a[1].v, false, bf.v,
                                                            (short)0, acc[1][t], false, false);
      }
    }
  }

#pragma unroll
  for (int i = 0; i < 2; ++i)
#pragma unroll
    for (int t = 0; t < 4; ++t) {
      int n = col0 + t * 16 + lr;
      float bv = bias ? bias[n] : 0.f;
#pragma unroll
      for (int r = 0; r < 8; ++r) {
        int m = row0 + i * 16 + r + 8 * g;
        if (m < M) {
          float v = acc[i][t][r] + bv;
          if (relu) v = fmaxf(v, 0.f);
          C[(size_t)m * N + n] = v;
        }
      }
    }
}

// ---------------------------------------------------------------------------
// Fused attention: softmax(Q K^T / 8) V.  One wave per (batch*head, 16-query
// tile), online softmax over 32-key blocks.  FULL template path has no masks
// and unguarded K/V loads; the single tail block takes the masked path.
// ---------------------------------------------------------------------------
template<bool FULL>
__device__ __forceinline__ void attn_step(
    const float* __restrict__ K, const float* __restrict__ V,
    int kb, int Lk, const BF16x16* qa, v8f* o,
    float* m_run, float* l_run, unsigned short* pls,
    int lane, int lr, int g)
{
  v8f s0 = Z8, s1 = Z8;
#pragma unroll
  for (int s2 = 0; s2 < 2; ++s2) {
    BF16x16 b0, b1;
    // One base address per fragment; kmap(j,g) folds into the VMEM immediate offset.
    const float* Kp = K + (size_t)(kb + lr) * Hs + s2 * 32;
#pragma unroll
    for (int j = 0; j < 16; ++j) {
      if (FULL) {
        b0.u[j] = f2bf(Kp[kmap(j, g)]);
        b1.u[j] = f2bf(Kp[16 * Hs + kmap(j, g)]);
      } else {
        b0.u[j] = ((kb + lr)      < Lk) ? f2bf(Kp[kmap(j, g)])           : (unsigned short)0;
        b1.u[j] = ((kb + 16 + lr) < Lk) ? f2bf(Kp[16 * Hs + kmap(j, g)]) : (unsigned short)0;
      }
    }
    s0 = __builtin_amdgcn_wmma_f32_16x16x32_bf16(false, qa[s2].v, false, b0.v,
                                                 (short)0, s0, false, false);
    s1 = __builtin_amdgcn_wmma_f32_16x16x32_bf16(false, qa[s2].v, false, b1.v,
                                                 (short)0, s1, false, false);
  }

  float mx[8];
#pragma unroll
  for (int r = 0; r < 8; ++r) {
    if (!FULL) {
      if ((kb + lr) >= Lk)      s0[r] = -1e30f;
      if ((kb + 16 + lr) >= Lk) s1[r] = -1e30f;
    }
    mx[r] = fmaxf(s0[r], s1[r]);
  }
#pragma unroll
  for (int offx = 1; offx < 16; offx <<= 1)
#pragma unroll
    for (int r = 0; r < 8; ++r)
      mx[r] = fmaxf(mx[r], __shfl_xor(mx[r], offx, 32));

  float p0[8], p1[8], rs[8];
#pragma unroll
  for (int r = 0; r < 8; ++r) {
    float mn = fmaxf(m_run[r], mx[r]);
    float alpha = __expf(m_run[r] - mn);
    m_run[r] = mn;
    l_run[r] *= alpha;
#pragma unroll
    for (int t = 0; t < 4; ++t) o[t][r] *= alpha;
    p0[r] = __expf(s0[r] - mn);
    p1[r] = __expf(s1[r] - mn);
    rs[r] = p0[r] + p1[r];
  }
#pragma unroll
  for (int offx = 1; offx < 16; offx <<= 1)
#pragma unroll
    for (int r = 0; r < 8; ++r)
      rs[r] += __shfl_xor(rs[r], offx, 32);
#pragma unroll
  for (int r = 0; r < 8; ++r) l_run[r] += rs[r];

  // C-layout P -> LDS -> A-layout fragment.
#pragma unroll
  for (int r = 0; r < 8; ++r) {
    pls[(r + 8 * g) * 32 + lr]      = f2bf(p0[r]);
    pls[(r + 8 * g) * 32 + 16 + lr] = f2bf(p1[r]);
  }
  __syncthreads();
  BF16x16 pa;
#pragma unroll
  for (int j = 0; j < 16; ++j)
    pa.u[j] = pls[lr * 32 + kmap(j, g)];
  __syncthreads();

  // O += P (16x32) x V (32x64)
  if (FULL) {
    const float* Vp = V + (size_t)kb * Hs + lr;
#pragma unroll
    for (int t = 0; t < 4; ++t) {
      BF16x16 vb;
#pragma unroll
      for (int j = 0; j < 16; ++j)
        vb.u[j] = f2bf(Vp[kmap(j, g) * Hs + t * 16]);
      o[t] = __builtin_amdgcn_wmma_f32_16x16x32_bf16(false, pa.v, false, vb.v,
                                                     (short)0, o[t], false, false);
    }
  } else {
    const float* Vp = V + (size_t)kb * Hs + lr;
#pragma unroll
    for (int t = 0; t < 4; ++t) {
      BF16x16 vb;
#pragma unroll
      for (int j = 0; j < 16; ++j)
        vb.u[j] = (kb + kmap(j, g) < Lk) ? f2bf(Vp[kmap(j, g) * Hs + t * 16]) : (unsigned short)0;
      o[t] = __builtin_amdgcn_wmma_f32_16x16x32_bf16(false, pa.v, false, vb.v,
                                                     (short)0, o[t], false, false);
    }
  }
}

__global__ __launch_bounds__(32) void attn_kernel(
    const float* __restrict__ Qb, const float* __restrict__ Kb,
    const float* __restrict__ Vb, float* __restrict__ Ob, int Lq, int Lk)
{
  __shared__ unsigned short pls[16 * 32];   // P tile staging (bf16)
  int lane = threadIdx.x & 31;
  int lr = lane & 15, g = lane >> 4;
  int bh = blockIdx.y;
  int b = bh / NHs, hh = bh % NHs;
  size_t off = ((size_t)b * Ls) * Hs + (size_t)hh * DHs;
  const float* Q = Qb + off;
  const float* K = Kb + off;
  const float* V = Vb + off;
  float* O = Ob + off;
  int row0 = blockIdx.x * 16;
  const float scale = 0.125f;   // 1/sqrt(64)

  // Q fragments for the two 32-wide K-dim steps, scale folded in (float4 loads).
  BF16x16 qa[2];
  int qrow = row0 + lr; if (qrow >= Lq) qrow = Lq - 1;
  const float* Qr = Q + (size_t)qrow * Hs;
#pragma unroll
  for (int s2 = 0; s2 < 2; ++s2) {
    const float* p = Qr + s2 * 32 + 8 * g;
    pack8s(qa[s2], 0, ld4(p),      ld4(p + 4),  scale);
    pack8s(qa[s2], 4, ld4(p + 16), ld4(p + 20), scale);
  }

  v8f o[4];
#pragma unroll
  for (int t = 0; t < 4; ++t) o[t] = Z8;
  float m_run[8], l_run[8];
#pragma unroll
  for (int r = 0; r < 8; ++r) { m_run[r] = -1e30f; l_run[r] = 0.f; }

  int kb = 0;
  for (; kb + 32 <= Lk; kb += 32)
    attn_step<true >(K, V, kb, Lk, qa, o, m_run, l_run, pls, lane, lr, g);
  if (kb < Lk)
    attn_step<false>(K, V, kb, Lk, qa, o, m_run, l_run, pls, lane, lr, g);

#pragma unroll
  for (int r = 0; r < 8; ++r) {
    float inv = 1.f / l_run[r];
    int m = row0 + r + 8 * g;
    if (m < Lq) {
#pragma unroll
      for (int t = 0; t < 4; ++t)
        O[(size_t)m * Hs + t * 16 + lr] = o[t][r] * inv;
    }
  }
}

// ---------------------------------------------------------------------------
// Elementwise / reduction helpers
// ---------------------------------------------------------------------------
__global__ __launch_bounds__(256) void input_proj_kernel(
    const float* __restrict__ x, const float* __restrict__ w,
    const float* __restrict__ bias, const float* __restrict__ pe,
    float* __restrict__ h)
{
  int row = blockIdx.x;
  int l = row % Ls;
  const float* xr = x + (size_t)row * DINs;
  float x0 = xr[0], x1 = xr[1], x2 = xr[2];   // channels 3: zeroed in encoder
  for (int c = threadIdx.x; c < Hs; c += 256) {
    float v = bias[c] + x0 * w[0 * Hs + c] + x1 * w[1 * Hs + c] + x2 * w[2 * Hs + c];
    h[(size_t)row * Hs + c] = v + pe[(size_t)l * Hs + c];
  }
}

__global__ __launch_bounds__(256) void query_init_kernel(
    const float* __restrict__ query, const float* __restrict__ pe,
    float* __restrict__ h)
{
  int row = blockIdx.x;
  int l = row % Ls;
  for (int c = threadIdx.x; c < Hs; c += 256)
    h[(size_t)row * Hs + c] = query[(size_t)l * Hs + c] + pe[(size_t)l * Hs + c];
}

__global__ __launch_bounds__(256) void add_ln_kernel(
    float* __restrict__ h, const float* __restrict__ dlt,
    const float* __restrict__ gamma, const float* __restrict__ beta)
{
  __shared__ float red[256];
  int row = blockIdx.x, t = threadIdx.x;
  size_t base = (size_t)row * Hs;
  float a = h[base + t] + dlt[base + t];
  float b = h[base + 256 + t] + dlt[base + 256 + t];
  red[t] = a + b;
  __syncthreads();
  for (int s = 128; s > 0; s >>= 1) { if (t < s) red[t] += red[t + s]; __syncthreads(); }
  float mu = red[0] * (1.f / 512.f);
  __syncthreads();
  float da = a - mu, db = b - mu;
  red[t] = da * da + db * db;
  __syncthreads();
  for (int s = 128; s > 0; s >>= 1) { if (t < s) red[t] += red[t + s]; __syncthreads(); }
  float rstd = rsqrtf(red[0] * (1.f / 512.f) + 1e-5f);
  h[base + t]       = da * rstd * gamma[t]       + beta[t];
  h[base + 256 + t] = db * rstd * gamma[256 + t] + beta[256 + t];
}

__global__ __launch_bounds__(64) void vq_norm_kernel(
    const float* __restrict__ zE, float* __restrict__ zN)
{
  __shared__ float red[64];
  int row = blockIdx.x, t = threadIdx.x;
  float v = zE[(size_t)row * CDs + t];
  red[t] = v * v;
  __syncthreads();
  for (int s = 32; s > 0; s >>= 1) { if (t < s) red[t] += red[t + s]; __syncthreads(); }
  float den = fmaxf(sqrtf(red[0]), 1e-8f);
  zN[(size_t)row * CDs + t] = v / den;
}

__global__ __launch_bounds__(256) void vq_argmax_kernel(
    const float* __restrict__ dots, int* __restrict__ idx, float* __restrict__ idxf)
{
  __shared__ float bv[256];
  __shared__ int bi[256];
  int row = blockIdx.x, t = threadIdx.x;
  float best = -1e30f; int besti = 0;
  for (int c = t; c < Kcb; c += 256) {
    float v = dots[(size_t)row * Kcb + c];
    if (v > best) { best = v; besti = c; }
  }
  bv[t] = best; bi[t] = besti;
  __syncthreads();
  for (int s = 128; s > 0; s >>= 1) {
    if (t < s) {
      if (bv[t + s] > bv[t] || (bv[t + s] == bv[t] && bi[t + s] < bi[t])) {
        bv[t] = bv[t + s]; bi[t] = bi[t + s];
      }
    }
    __syncthreads();
  }
  if (t == 0) { idx[row] = bi[0]; idxf[row] = (float)bi[0]; }
}

__global__ __launch_bounds__(64) void vq_gather_kernel(
    const float* __restrict__ cb, const int* __restrict__ idx, float* __restrict__ zQ)
{
  int row = blockIdx.x, t = threadIdx.x;
  zQ[(size_t)row * CDs + t] = cb[(size_t)idx[row] * CDs + t];
}

__global__ __launch_bounds__(256) void zero_kernel(float* __restrict__ p, int n) {
  int i = blockIdx.x * 256 + threadIdx.x;
  if (i < n) p[i] = 0.f;
}

__global__ __launch_bounds__(256) void vq_hist_kernel(
    const int* __restrict__ idx, float* __restrict__ usage)
{
  int i = blockIdx.x * 256 + threadIdx.x;
  if (i < BL) atomicAdd(&usage[idx[i]], 1.f);
}

__global__ __launch_bounds__(256) void vq_stats_kernel(
    const float* __restrict__ usage, float* __restrict__ st)
{
  __shared__ float red[256];
  int t = threadIdx.x;
  float s = 0.f;
  for (int c = t; c < Kcb; c += 256) s += usage[c];
  red[t] = s; __syncthreads();
  for (int k = 128; k > 0; k >>= 1) { if (t < k) red[t] += red[t + k]; __syncthreads(); }
  float total = fmaxf(red[0], 1.f);
  __syncthreads();
  float ent = 0.f, dead = 0.f;
  for (int c = t; c < Kcb; c += 256) {
    float u = usage[c];
    float p = u / total;
    if (p > 0.f) ent += p * __logf(p);
    if (u == 0.f) dead += 1.f;
  }
  red[t] = ent; __syncthreads();
  for (int k = 128; k > 0; k >>= 1) { if (t < k) red[t] += red[t + k]; __syncthreads(); }
  float entT = red[0];
  __syncthreads();
  red[t] = dead; __syncthreads();
  for (int k = 128; k > 0; k >>= 1) { if (t < k) red[t] += red[t + k]; __syncthreads(); }
  if (t == 0) { st[0] = __expf(-entT); st[1] = red[0] / (float)Kcb; }
}

__global__ __launch_bounds__(256) void head_kernel(
    const float* __restrict__ hd,
    const float* __restrict__ wxyz, const float* __restrict__ bxyz,
    const float* __restrict__ wss,  const float* __restrict__ bss,
    float* __restrict__ out)
{
  __shared__ float sh[512];
  int row = blockIdx.x, t = threadIdx.x;
  sh[t]       = hd[(size_t)row * Hs + t];
  sh[t + 256] = hd[(size_t)row * Hs + 256 + t];
  __syncthreads();
  int w = t >> 5, lane = t & 31;
  if (w < 6) {
    const float* W = (w < 3) ? wxyz : wss;
    int c = (w < 3) ? w : (w - 3);
    float s = 0.f;
    for (int d = lane; d < Hs; d += 32) s += sh[d] * W[(size_t)d * 3 + c];
    for (int offx = 16; offx > 0; offx >>= 1) s += __shfl_xor(s, offx, 32);
    if (lane == 0) out[(size_t)row * 6 + w] = s + ((w < 3) ? bxyz[c] : bss[c]);
  }
}

// ---------------------------------------------------------------------------
// Host orchestration
// ---------------------------------------------------------------------------
static void launch_gemm(const float* A, const float* B, const float* bias,
                        float* C, int M, int N, int Kd, int transB, int relu,
                        hipStream_t s)
{
  dim3 grid((N + 255) / 256, (M + 31) / 32);
  gemm_wmma_kernel<<<grid, 128, 0, s>>>(A, B, bias, C, M, N, Kd, transB, relu);
}

extern "C" void kernel_launch(void* const* d_in, const int* in_sizes, int n_in,
                              void* d_out, int out_size, void* d_ws, size_t ws_size,
                              hipStream_t stream)
{
  (void)in_sizes; (void)n_in; (void)out_size; (void)ws_size;
  // jax pytree (sorted-key) flatten order of setup_inputs():
  //  0: x
  //  1: codebook
  //  2..105:  dec[0..3] x {ca.{k,o,q,v}.{b,w}, ff1.{b,w}, ff2.{b,w},
  //                        ln1.{b,g}, ln2.{b,g}, ln3.{b,g}, sa.{k,o,q,v}.{b,w}}
  //  106..169: enc[0..3] x {attn.{k,o,q,v}.{b,w}, ff1.{b,w}, ff2.{b,w},
  //                         ln1.{b,g}, ln2.{b,g}}
  //  170/171: from_code b/w   172/173: head_ss b/w   174/175: head_xyz b/w
  //  176/177: input_proj b/w  178: pos  179: query   180/181: to_code b/w
  const float* x        = (const float*)d_in[0];
  const float* codebook = (const float*)d_in[1];
  auto DEC = [&](int l, int o) { return (const float*)d_in[2 + 26 * l + o]; };
  auto ENC = [&](int l, int o) { return (const float*)d_in[106 + 16 * l + o]; };
  const float* from_b = (const float*)d_in[170];
  const float* from_w = (const float*)d_in[171];
  const float* hss_b  = (const float*)d_in[172];
  const float* hss_w  = (const float*)d_in[173];
  const float* hxyz_b = (const float*)d_in[174];
  const float* hxyz_w = (const float*)d_in[175];
  const float* ip_b   = (const float*)d_in[176];
  const float* ip_w   = (const float*)d_in[177];
  const float* pos    = (const float*)d_in[178];
  const float* query  = (const float*)d_in[179];
  const float* tc_b   = (const float*)d_in[180];
  const float* tc_w   = (const float*)d_in[181];

  float* out = (float*)d_out;
  float* o_recons = out;                   // BL*6
  float* o_idx    = out + (size_t)BL * 6;  // BL
  float* o_stats  = o_idx + BL;            // 2

  // Workspace layout
  float* ws   = (float*)d_ws;
  float* h    = ws;                              // BL*Hs
  float* t1   = h   + (size_t)BL * Hs;           // BL*Hs
  float* ao   = t1  + (size_t)BL * Hs;           // BL*Hs
  float* qh   = ao  + (size_t)BL * Hs;           // BL*Hs
  float* kh   = qh  + (size_t)BL * Hs;           // BL*Hs
  float* vh   = kh  + (size_t)BL * Hs;           // BL*Hs
  float* mem  = vh  + (size_t)BL * Hs;           // BL*Hs
  float* ff   = mem + (size_t)BL * Hs;           // BL*FFs (also VQ dots BL*Kcb)
  float* zE   = ff  + (size_t)BL * FFs;          // BL*CDs
  float* zQ   = zE  + (size_t)BL * CDs;          // BL*CDs
  float* usage = zQ + (size_t)BL * CDs;          // Kcb
  int*   idx  = (int*)(usage + Kcb);             // BL ints

  dim3 agrid((Ls + 15) / 16, Bb * NHs);

  // ----- Encoder -----
  input_proj_kernel<<<BL, 256, 0, stream>>>(x, ip_w, ip_b, pos, h);
  for (int l = 0; l < NLs; ++l) {
    launch_gemm(h, ENC(l, 5), ENC(l, 4), qh, BL, Hs, Hs, 0, 0, stream);  // attn.q
    launch_gemm(h, ENC(l, 1), ENC(l, 0), kh, BL, Hs, Hs, 0, 0, stream);  // attn.k
    launch_gemm(h, ENC(l, 7), ENC(l, 6), vh, BL, Hs, Hs, 0, 0, stream);  // attn.v
    attn_kernel<<<agrid, 32, 0, stream>>>(qh, kh, vh, ao, Ls, Ls);
    launch_gemm(ao, ENC(l, 3), ENC(l, 2), t1, BL, Hs, Hs, 0, 0, stream); // attn.o
    add_ln_kernel<<<BL, 256, 0, stream>>>(h, t1, ENC(l, 13), ENC(l, 12)); // ln1
    launch_gemm(h, ENC(l, 9), ENC(l, 8), ff, BL, FFs, Hs, 0, 1, stream);  // ff1+relu
    launch_gemm(ff, ENC(l, 11), ENC(l, 10), t1, BL, Hs, FFs, 0, 0, stream); // ff2
    add_ln_kernel<<<BL, 256, 0, stream>>>(h, t1, ENC(l, 15), ENC(l, 14)); // ln2
  }

  // ----- VQ -----
  launch_gemm(h, tc_w, tc_b, zE, BL, CDs, Hs, 0, 0, stream);
  vq_norm_kernel<<<BL, 64, 0, stream>>>(zE, t1);
  launch_gemm(t1, codebook, nullptr, ff, BL, Kcb, CDs, 1, 0, stream);  // dots
  vq_argmax_kernel<<<BL, 256, 0, stream>>>(ff, idx, o_idx);
  vq_gather_kernel<<<BL, 64, 0, stream>>>(codebook, idx, zQ);
  zero_kernel<<<(Kcb + 255) / 256, 256, 0, stream>>>(usage, Kcb);
  vq_hist_kernel<<<(BL + 255) / 256, 256, 0, stream>>>(idx, usage);
  vq_stats_kernel<<<1, 256, 0, stream>>>(usage, o_stats);
  launch_gemm(zQ, from_w, from_b, mem, BL, Hs, CDs, 0, 0, stream);

  // ----- Decoder -----
  query_init_kernel<<<BL, 256, 0, stream>>>(query, pos, h);
  for (int l = 0; l < NLs; ++l) {
    // self-attention
    launch_gemm(h, DEC(l, 23), DEC(l, 22), qh, BL, Hs, Hs, 0, 0, stream); // sa.q
    launch_gemm(h, DEC(l, 19), DEC(l, 18), kh, BL, Hs, Hs, 0, 0, stream); // sa.k
    launch_gemm(h, DEC(l, 25), DEC(l, 24), vh, BL, Hs, Hs, 0, 0, stream); // sa.v
    attn_kernel<<<agrid, 32, 0, stream>>>(qh, kh, vh, ao, Ls, Ls);
    launch_gemm(ao, DEC(l, 21), DEC(l, 20), t1, BL, Hs, Hs, 0, 0, stream); // sa.o
    add_ln_kernel<<<BL, 256, 0, stream>>>(h, t1, DEC(l, 13), DEC(l, 12)); // ln1
    // cross-attention
    launch_gemm(h,   DEC(l, 5), DEC(l, 4), qh, BL, Hs, Hs, 0, 0, stream); // ca.q
    launch_gemm(mem, DEC(l, 1), DEC(l, 0), kh, BL, Hs, Hs, 0, 0, stream); // ca.k
    launch_gemm(mem, DEC(l, 7), DEC(l, 6), vh, BL, Hs, Hs, 0, 0, stream); // ca.v
    attn_kernel<<<agrid, 32, 0, stream>>>(qh, kh, vh, ao, Ls, Ls);
    launch_gemm(ao, DEC(l, 3), DEC(l, 2), t1, BL, Hs, Hs, 0, 0, stream);  // ca.o
    add_ln_kernel<<<BL, 256, 0, stream>>>(h, t1, DEC(l, 15), DEC(l, 14)); // ln2
    // FFN
    launch_gemm(h, DEC(l, 9), DEC(l, 8), ff, BL, FFs, Hs, 0, 1, stream);
    launch_gemm(ff, DEC(l, 11), DEC(l, 10), t1, BL, Hs, FFs, 0, 0, stream);
    add_ln_kernel<<<BL, 256, 0, stream>>>(h, t1, DEC(l, 17), DEC(l, 16)); // ln3
  }

  // ----- Heads -----
  head_kernel<<<BL, 256, 0, stream>>>(h, hxyz_w, hxyz_b, hss_w, hss_b, o_recons);
}
